// Decoder_26963804685169
// MI455X (gfx1250) — matrix operand order repaired and checked by reading, not compiled, not run on hardware
//
#include <hip/hip_runtime.h>
#include <hip/hip_bf16.h>
#include <math.h>

typedef __bf16 bf16_t;
typedef bf16_t v16bf  __attribute__((ext_vector_type(16)));
typedef bf16_t bf16x8 __attribute__((ext_vector_type(8)));
typedef float  v8f    __attribute__((ext_vector_type(8)));

#define NB 64
#define NS 80
#define NTS 39          // T-1 decode steps
#define NH 512
#define NE 1024
#define NV 32000
#define NROWS (NB*NTS)  // 2496
#define NCHUNK 10       // vocab N split: 2000 tiles = 10 chunks * 200 tiles

// ---------------------------------------------------------------------------
// helpers
// ---------------------------------------------------------------------------
__device__ __forceinline__ void split_bf16(float x, bf16_t* hi, bf16_t* lo) {
  bf16_t h = (bf16_t)x;
  *hi = h;
  *lo = (bf16_t)(x - (float)h);
}

// A fragment (16x32 bf16): lane 0-15 -> K {0..7,16..23}, lane 16-31 -> +8
__device__ __forceinline__ v16bf load_a_frag(const bf16_t* rowPtr, int sel) {
  bf16x8 p0 = *(const bf16x8*)(rowPtr + sel * 8);
  bf16x8 p1 = *(const bf16x8*)(rowPtr + 16 + sel * 8);
  return __builtin_shufflevector(p0, p1, 0,1,2,3,4,5,6,7,8,9,10,11,12,13,14,15);
}
// B fragment (32x16 bf16): lane 0-15 -> K {0..7,16..23}, lane 16-31 -> +8
__device__ __forceinline__ v16bf load_b_frag(const bf16_t* rowPtr, int sel) {
  bf16x8 p0 = *(const bf16x8*)(rowPtr + sel * 8);
  bf16x8 p1 = *(const bf16x8*)(rowPtr + 16 + sel * 8);
  return __builtin_shufflevector(p0, p1, 0,1,2,3,4,5,6,7,8,9,10,11,12,13,14,15);
}

// ---------------------------------------------------------------------------
// one-time prep kernels
// ---------------------------------------------------------------------------
__global__ void convert_split_kernel(const float* __restrict__ src,
                                     bf16_t* __restrict__ hi, bf16_t* __restrict__ lo,
                                     long n) {
  long i = (long)blockIdx.x * 256 + threadIdx.x;
  if (i >= n) return;
  split_bf16(src[i], &hi[i], &lo[i]);
}

// dst[n*512+k] = src[(rowOff+k)*512 + n]   (512x512 transpose + split)
__global__ void transpose_split_kernel(const float* __restrict__ src, int rowOff,
                                       bf16_t* __restrict__ hi, bf16_t* __restrict__ lo) {
  int i = blockIdx.x * 256 + threadIdx.x;
  if (i >= 512 * 512) return;
  int n = i >> 9, k = i & 511;
  split_bf16(src[(size_t)(rowOff + k) * 512 + n], &hi[i], &lo[i]);
}

// xs[(t*64+b)*1024+e] = emb[targets[b*40+t]*1024+e]  (split bf16)
__global__ void gather_embed_kernel(const float* __restrict__ emb,
                                    const int* __restrict__ targets,
                                    bf16_t* __restrict__ hi, bf16_t* __restrict__ lo) {
  long i = (long)blockIdx.x * 256 + threadIdx.x;
  if (i >= (long)NTS * NB * NE) return;
  int  e  = (int)(i & (NE - 1));
  long tb = i >> 10;
  int  b  = (int)(tb % NB);
  int  t  = (int)(tb / NB);
  int tok = targets[b * 40 + t];
  split_bf16(emb[(size_t)tok * NE + e], &hi[i], &lo[i]);
}

__global__ void h_init_kernel(const float* __restrict__ hidden, float* __restrict__ h,
                              bf16_t* __restrict__ hhi, bf16_t* __restrict__ hlo) {
  int i = blockIdx.x * 256 + threadIdx.x;
  if (i >= NB * NH) return;
  float x = hidden[i];
  h[i] = x;
  split_bf16(x, &hhi[i], &hlo[i]);
}

// ---------------------------------------------------------------------------
// generic split-precision bf16 WMMA GEMM:  C[M,N] = A[M,K] * B^T + bias
// A may come from two K-segments (seg0 then seg1) -> two straight-line loops,
// no per-k-step branch. B stored (n,k)-major, ldB. One wave = one 16x16 tile.
// ---------------------------------------------------------------------------
__global__ __launch_bounds__(256) void gemm_split_kernel(
    const bf16_t* __restrict__ A0h, const bf16_t* __restrict__ A0l, int ldA0, int K0,
    const bf16_t* __restrict__ A1h, const bf16_t* __restrict__ A1l, int ldA1, int K1,
    const bf16_t* __restrict__ Bh,  const bf16_t* __restrict__ Bl,  int ldB,
    const float* __restrict__ bias, float* __restrict__ C, int ldC,
    int Mtiles, int Ntiles)
{
  int lane = threadIdx.x & 31;
  int wave = (blockIdx.x * blockDim.x + threadIdx.x) >> 5;
  if (wave >= Mtiles * Ntiles) return;         // wave-uniform: EXEC stays all-1s
  int mt = wave / Ntiles, nt = wave - mt * Ntiles;
  int r = lane & 15, sel = lane >> 4;
  v8f c = {0.f, 0.f, 0.f, 0.f, 0.f, 0.f, 0.f, 0.f};

  const bf16_t* bp  = Bh + (size_t)(nt * 16 + r) * ldB;
  const bf16_t* blp = Bl + (size_t)(nt * 16 + r) * ldB;

  const bf16_t* ap  = A0h + (size_t)(mt * 16 + r) * ldA0;
  const bf16_t* alp = A0l + (size_t)(mt * 16 + r) * ldA0;
  for (int kb = 0; kb < K0; kb += 32) {
    v16bf aH = load_a_frag(ap + kb,  sel);
    v16bf aL = load_a_frag(alp + kb, sel);
    v16bf bH = load_b_frag(bp + kb,  sel);
    v16bf bL = load_b_frag(blp + kb, sel);
    c = __builtin_amdgcn_wmma_f32_16x16x32_bf16(false, aH, false, bH, (short)0, c, false, false);
    c = __builtin_amdgcn_wmma_f32_16x16x32_bf16(false, aL, false, bH, (short)0, c, false, false);
    c = __builtin_amdgcn_wmma_f32_16x16x32_bf16(false, aH, false, bL, (short)0, c, false, false);
  }
  if (K1 > 0) {
    ap  = A1h + (size_t)(mt * 16 + r) * ldA1;
    alp = A1l + (size_t)(mt * 16 + r) * ldA1;
    for (int kb = 0; kb < K1; kb += 32) {
      v16bf aH = load_a_frag(ap + kb,  sel);
      v16bf aL = load_a_frag(alp + kb, sel);
      v16bf bH = load_b_frag(bp + K0 + kb,  sel);
      v16bf bL = load_b_frag(blp + K0 + kb, sel);
      c = __builtin_amdgcn_wmma_f32_16x16x32_bf16(false, aH, false, bH, (short)0, c, false, false);
      c = __builtin_amdgcn_wmma_f32_16x16x32_bf16(false, aL, false, bH, (short)0, c, false, false);
      c = __builtin_amdgcn_wmma_f32_16x16x32_bf16(false, aH, false, bL, (short)0, c, false, false);
    }
  }
  int ncol = nt * 16 + r;
  float bv = bias ? bias[ncol] : 0.0f;
#pragma unroll
  for (int i = 0; i < 8; ++i) {
    int m = mt * 16 + sel * 8 + i;
    C[(size_t)m * ldC + ncol] = c[i] + bv;
  }
}

// ---------------------------------------------------------------------------
// attention: per-batch block. hproj = h @ Wa[:512] (VALU fp32, Wa L2-resident),
// scores -> softmax(S=80) -> ctx, emitted as bf16 hi/lo for the GRU GEMM.
// enc_proj already contains enc @ Wa[512:] + ba (hoisted out of the scan).
// ---------------------------------------------------------------------------
__global__ __launch_bounds__(256) void attention_kernel(
    const float* __restrict__ h, const float* __restrict__ Wa,
    const float* __restrict__ wf,
    const float* __restrict__ enc, const float* __restrict__ enc_proj,
    bf16_t* __restrict__ ctx_hi, bf16_t* __restrict__ ctx_lo)
{
  int b = blockIdx.x, tid = threadIdx.x;
  __shared__ float hs[NH];
  __shared__ float hproj[NH];
  __shared__ float sc[NS];
  __shared__ float attnv[NS];
  __shared__ float stat[2];
  for (int j = tid; j < NH; j += 256) hs[j] = h[b * NH + j];
  __syncthreads();
  for (int j = tid; j < NH; j += 256) {
    float acc = 0.f;
    for (int k = 0; k < NH; ++k) acc = fmaf(hs[k], Wa[(size_t)k * NH + j], acc);
    hproj[j] = acc;
  }
  __syncthreads();
  int w = tid >> 5, lane = tid & 31;
  for (int s = w; s < NS; s += 8) {
    const float* ep = enc_proj + ((size_t)b * NS + s) * NH;
    float acc = 0.f;
    for (int hh = lane; hh < NH; hh += 32) {
      float e = fmaxf(hproj[hh] + ep[hh], 0.f);     // relu
      acc = fmaf(e, wf[hh], acc);
    }
#pragma unroll
    for (int off = 16; off; off >>= 1) acc += __shfl_down(acc, off, 32);
    if (lane == 0) sc[s] = acc;
  }
  __syncthreads();
  if (tid == 0) {
    float mx = sc[0];
    for (int s = 1; s < NS; ++s) mx = fmaxf(mx, sc[s]);
    float sum = 0.f;
    for (int s = 0; s < NS; ++s) sum += __expf(sc[s] - mx);
    stat[0] = mx; stat[1] = sum;
  }
  __syncthreads();
  if (tid < NS) attnv[tid] = __expf(sc[tid] - stat[0]) / stat[1];
  __syncthreads();
  for (int j = tid; j < NH; j += 256) {
    float acc = 0.f;
    const float* eb = enc + (size_t)b * NS * NH + j;
    for (int s = 0; s < NS; ++s) acc = fmaf(attnv[s], eb[(size_t)s * NH], acc);
    split_bf16(acc, &ctx_hi[b * NH + j], &ctx_lo[b * NH + j]);
  }
}

// ---------------------------------------------------------------------------
// GRU gate nonlinearity + state update (+ emit outputs row in fp32 and bf16)
// ---------------------------------------------------------------------------
__global__ void gru_gate_kernel(const float* __restrict__ gi, const float* __restrict__ gh,
                                float* __restrict__ h, bf16_t* __restrict__ hhi,
                                bf16_t* __restrict__ hlo,
                                float* __restrict__ outputs, bf16_t* __restrict__ ohi,
                                bf16_t* __restrict__ olo, int t)
{
  int idx = blockIdx.x * 256 + threadIdx.x;
  if (idx >= NB * NH) return;
  int b = idx >> 9, j = idx & 511;
  const float* gib = gi + (size_t)b * 1536;
  const float* ghb = gh + (size_t)b * 1536;
  float rg = 1.f / (1.f + __expf(-(gib[j]        + ghb[j])));
  float zg = 1.f / (1.f + __expf(-(gib[512 + j]  + ghb[512 + j])));
  float ng = tanhf(gib[1024 + j] + rg * ghb[1024 + j]);
  float hv = h[idx];
  float hn = (1.f - zg) * ng + zg * hv;
  h[idx] = hn;
  bf16_t hb = (bf16_t)hn;
  bf16_t lb = (bf16_t)(hn - (float)hb);
  hhi[idx] = hb; hlo[idx] = lb;
  size_t orow = ((size_t)b * NTS + t) * NH + j;
  outputs[orow] = hn; ohi[orow] = hb; olo[orow] = lb;
}

// ---------------------------------------------------------------------------
// vocab projection: 64 rows per WG; A tile staged into 128KB LDS via CDNA5
// GLOBAL_LOAD_ASYNC_TO_LDS_B128 (ASYNCcnt), then s_wait_asynccnt + barrier.
// 10 N-chunks; each wave computes TWO adjacent 16x16 N-tiles per iteration so
// every A fragment (hi+lo) feeds 6 WMMAs. Fused online logsumexp epilogue.
// grid = 39 rowBlocks * 10 chunks, 256 threads (8 waves).
// ---------------------------------------------------------------------------
__global__ __launch_bounds__(256) void vocab_kernel(
    const bf16_t* __restrict__ Ah, const bf16_t* __restrict__ Al,   // (2496,512)
    const bf16_t* __restrict__ Bh, const bf16_t* __restrict__ Bl,   // (32000,512)
    const float* __restrict__ bo,
    float* __restrict__ out, float* __restrict__ partM, float* __restrict__ partS)
{
  extern __shared__ char smem[];
  bf16_t* lAh = (bf16_t*)smem;                       // 64*512 bf16 (64KB)
  bf16_t* lAl = (bf16_t*)(smem + 64 * 512 * 2);      // 64*512 bf16 (64KB)
  __shared__ float redm[8][16];
  __shared__ float reds[8][16];

  int tid = threadIdx.x;
  int rowBlock = blockIdx.x / NCHUNK;
  int chunk    = blockIdx.x % NCHUNK;
  int rowBase  = rowBlock * 64;

  // ---- async memory -> LDS staging (per-lane 16B DMA, no VGPR round-trip) ----
  {
    unsigned ldsH = (unsigned)(uintptr_t)lAh;
    unsigned ldsL = (unsigned)(uintptr_t)lAl;
    const char* gH = (const char*)(Ah + (size_t)rowBase * 512);
    const char* gL = (const char*)(Al + (size_t)rowBase * 512);
    for (int i = tid; i < 64 * 512 / 8; i += 256) {
      unsigned dH = ldsH + i * 16;
      unsigned dL = ldsL + i * 16;
      const char* sH = gH + (size_t)i * 16;
      const char* sL = gL + (size_t)i * 16;
      asm volatile("global_load_async_to_lds_b128 %0, %1, off"
                   :: "v"(dH), "v"(sH) : "memory");
      asm volatile("global_load_async_to_lds_b128 %0, %1, off"
                   :: "v"(dL), "v"(sL) : "memory");
    }
    asm volatile("s_wait_asynccnt 0x0" ::: "memory");
  }
  __syncthreads();

  int lane = tid & 31, w = tid >> 5;
  int r = lane & 15, sel = lane >> 4;
  int mt = w & 3;            // which 16-row tile within the 64 rows
  int nh = w >> 2;           // N phase: waves alternate pairs of tiles

  float m8[8], s8[8];
#pragma unroll
  for (int i = 0; i < 8; ++i) { m8[i] = -INFINITY; s8[i] = 0.f; }

  const bf16_t* ap  = lAh + (size_t)(mt * 16 + r) * 512;
  const bf16_t* alp = lAl + (size_t)(mt * 16 + r) * 512;
  int ntStart = chunk * 200;                  // 2000 N-tiles / 10 chunks
  for (int it = 0; it < 50; ++it) {
    int nt = ntStart + nh * 2 + it * 4;       // this wave: tiles nt and nt+1
    v8f c0 = {0.f, 0.f, 0.f, 0.f, 0.f, 0.f, 0.f, 0.f};
    v8f c1 = {0.f, 0.f, 0.f, 0.f, 0.f, 0.f, 0.f, 0.f};
    const bf16_t* bp0  = Bh + (size_t)(nt * 16 + r) * 512;
    const bf16_t* blp0 = Bl + (size_t)(nt * 16 + r) * 512;
    const bf16_t* bp1  = bp0 + (size_t)16 * 512;
    const bf16_t* blp1 = blp0 + (size_t)16 * 512;
#pragma unroll 4
    for (int kb = 0; kb < 512; kb += 32) {
      v16bf aH  = load_a_frag(ap + kb,  sel);
      v16bf aL  = load_a_frag(alp + kb, sel);
      v16bf bH0 = load_b_frag(bp0 + kb,  sel);
      v16bf bL0 = load_b_frag(blp0 + kb, sel);
      v16bf bH1 = load_b_frag(bp1 + kb,  sel);
      v16bf bL1 = load_b_frag(blp1 + kb, sel);
      c0 = __builtin_amdgcn_wmma_f32_16x16x32_bf16(false, aH, false, bH0, (short)0, c0, false, false);
      c0 = __builtin_amdgcn_wmma_f32_16x16x32_bf16(false, aL, false, bH0, (short)0, c0, false, false);
      c0 = __builtin_amdgcn_wmma_f32_16x16x32_bf16(false, aH, false, bL0, (short)0, c0, false, false);
      c1 = __builtin_amdgcn_wmma_f32_16x16x32_bf16(false, aH, false, bH1, (short)0, c1, false, false);
      c1 = __builtin_amdgcn_wmma_f32_16x16x32_bf16(false, aL, false, bH1, (short)0, c1, false, false);
      c1 = __builtin_amdgcn_wmma_f32_16x16x32_bf16(false, aH, false, bL1, (short)0, c1, false, false);
    }
    int ncol0 = nt * 16 + r;
    float bv0 = bo[ncol0];
    float bv1 = bo[ncol0 + 16];
    int mBase = rowBase + mt * 16 + sel * 8;
#pragma unroll
    for (int i = 0; i < 8; ++i) {
      float x0 = c0[i] + bv0;
      float x1 = c1[i] + bv1;
      out[(size_t)(mBase + i) * NV + ncol0]      = x0;
      out[(size_t)(mBase + i) * NV + ncol0 + 16] = x1;
      float nm = fmaxf(m8[i], fmaxf(x0, x1));
      s8[i] = s8[i] * __expf(m8[i] - nm) + __expf(x0 - nm) + __expf(x1 - nm);
      m8[i] = nm;
    }
  }
  // reduce across the 16 lanes holding the same rows (xor < 16 stays in half)
#pragma unroll
  for (int off = 1; off < 16; off <<= 1) {
#pragma unroll
    for (int i = 0; i < 8; ++i) {
      float om = __shfl_xor(m8[i], off, 32);
      float os = __shfl_xor(s8[i], off, 32);
      float nm = fmaxf(m8[i], om);
      s8[i] = s8[i] * __expf(m8[i] - nm) + os * __expf(om - nm);
      m8[i] = nm;
    }
  }
  if (r == 0) {
#pragma unroll
    for (int i = 0; i < 8; ++i) { redm[w][sel * 8 + i] = m8[i]; reds[w][sel * 8 + i] = s8[i]; }
  }
  __syncthreads();
  if (tid < 64) {   // combine the two waves (nh=0/1) per row, store chunk partial
    int mt2 = tid >> 4, sub = tid & 15;
    float m0 = redm[mt2][sub],     s0 = reds[mt2][sub];
    float m1 = redm[mt2 + 4][sub], s1 = reds[mt2 + 4][sub];
    float nm = fmaxf(m0, m1);
    float s  = s0 * __expf(m0 - nm) + s1 * __expf(m1 - nm);
    partM[(size_t)(rowBase + tid) * NCHUNK + chunk] = nm;
    partS[(size_t)(rowBase + tid) * NCHUNK + chunk] = s;
  }
}

__global__ void combine_stats_kernel(const float* __restrict__ pm,
                                     const float* __restrict__ ps,
                                     float* __restrict__ rowStats) {
  int row = blockIdx.x * 256 + threadIdx.x;
  if (row >= NROWS) return;
  float m = -INFINITY, s = 0.f;
  for (int c = 0; c < NCHUNK; ++c) {
    float cm = pm[(size_t)row * NCHUNK + c], cs = ps[(size_t)row * NCHUNK + c];
    float nm = fmaxf(m, cm);
    s = s * __expf(m - nm) + cs * __expf(cm - nm);
    m = nm;
  }
  rowStats[row] = m + __logf(s);
}

__global__ void logsoftmax_fix_kernel(float* __restrict__ out,
                                      const float* __restrict__ rowStats) {
  size_t i4 = (size_t)blockIdx.x * 256 + threadIdx.x;
  size_t total4 = (size_t)NROWS * NV / 4;
  if (i4 >= total4) return;
  float4 v = ((float4*)out)[i4];
  int row = (int)((i4 * 4) / NV);
  float off = rowStats[row];
  v.x -= off; v.y -= off; v.z -= off; v.w -= off;
  ((float4*)out)[i4] = v;
}

__global__ void argmax_kernel(const float* __restrict__ outputs, float* __restrict__ dst) {
  __shared__ float bv[128];
  __shared__ int   bi[128];
  int row = blockIdx.x, tid = threadIdx.x;
  float best = -INFINITY; int bidx = 0;
  for (int j = tid; j < NH; j += 128) {
    float v = outputs[(size_t)row * NH + j];
    if (v > best) { best = v; bidx = j; }
  }
  bv[tid] = best; bi[tid] = bidx;
  __syncthreads();
  for (int s = 64; s; s >>= 1) {
    if (tid < s) {
      if (bv[tid + s] > bv[tid] || (bv[tid + s] == bv[tid] && bi[tid + s] < bi[tid])) {
        bv[tid] = bv[tid + s]; bi[tid] = bi[tid + s];
      }
    }
    __syncthreads();
  }
  if (tid == 0) dst[row] = (float)bi[0];
}

// ---------------------------------------------------------------------------
extern "C" void kernel_launch(void* const* d_in, const int* in_sizes, int n_in,
                              void* d_out, int out_size, void* d_ws, size_t ws_size,
                              hipStream_t stream)
{
  (void)in_sizes; (void)n_in; (void)out_size; (void)ws_size;
  const float* hidden  = (const float*)d_in[0];
  const float* enc     = (const float*)d_in[1];
  const int*   targets = (const int*)  d_in[2];
  const float* emb     = (const float*)d_in[3];
  const float* Wa      = (const float*)d_in[4];
  const float* ba      = (const float*)d_in[5];
  const float* wf      = (const float*)d_in[6];
  const float* W_ih    = (const float*)d_in[7];
  const float* W_hh    = (const float*)d_in[8];
  const float* b_ih    = (const float*)d_in[9];
  const float* b_hh    = (const float*)d_in[10];
  const float* Wo      = (const float*)d_in[11];
  const float* bo      = (const float*)d_in[12];
  float* logp = (float*)d_out;
  float* amax = logp + (size_t)NROWS * NV;

  // ---- carve workspace ----
  char* p = (char*)d_ws;
  auto carve = [&](size_t bytes) -> char* {
    char* q = p; p += (bytes + 255) & ~(size_t)255; return q;
  };
  bf16_t* WihH   = (bf16_t*)carve((size_t)1536 * 1536 * 2);
  bf16_t* WihL   = (bf16_t*)carve((size_t)1536 * 1536 * 2);
  bf16_t* WhhH   = (bf16_t*)carve((size_t)1536 * 512 * 2);
  bf16_t* WhhL   = (bf16_t*)carve((size_t)1536 * 512 * 2);
  bf16_t* WoH    = (bf16_t*)carve((size_t)NV * NH * 2);
  bf16_t* WoL    = (bf16_t*)carve((size_t)NV * NH * 2);
  bf16_t* WaEncTH= (bf16_t*)carve((size_t)512 * 512 * 2);
  bf16_t* WaEncTL= (bf16_t*)carve((size_t)512 * 512 * 2);
  bf16_t* xsH    = (bf16_t*)carve((size_t)NTS * NB * NE * 2);
  bf16_t* xsL    = (bf16_t*)carve((size_t)NTS * NB * NE * 2);
  bf16_t* encH   = (bf16_t*)carve((size_t)NB * NS * NH * 2);
  bf16_t* encL   = (bf16_t*)carve((size_t)NB * NS * NH * 2);
  float*  encProj= (float*) carve((size_t)NB * NS * NH * 4);
  float*  hbuf   = (float*) carve((size_t)NB * NH * 4);
  bf16_t* hH     = (bf16_t*)carve((size_t)NB * NH * 2);
  bf16_t* hL     = (bf16_t*)carve((size_t)NB * NH * 2);
  bf16_t* ctxH   = (bf16_t*)carve((size_t)NB * NH * 2);
  bf16_t* ctxL   = (bf16_t*)carve((size_t)NB * NH * 2);
  float*  gi     = (float*) carve((size_t)NB * 1536 * 4);
  float*  gh     = (float*) carve((size_t)NB * 1536 * 4);
  float*  outputs= (float*) carve((size_t)NROWS * NH * 4);
  bf16_t* outH   = (bf16_t*)carve((size_t)NROWS * NH * 2);
  bf16_t* outL   = (bf16_t*)carve((size_t)NROWS * NH * 2);
  float*  partM  = (float*) carve((size_t)NROWS * NCHUNK * 4);
  float*  partS  = (float*) carve((size_t)NROWS * NCHUNK * 4);
  float*  rowStats=(float*) carve((size_t)NROWS * 4);

  auto cdiv = [](long n, long d) { return (int)((n + d - 1) / d); };
  auto launch_gemm = [&](const bf16_t* A0h, const bf16_t* A0l, int ld0, int K0,
                         const bf16_t* A1h, const bf16_t* A1l, int ld1, int K1,
                         const bf16_t* Bh, const bf16_t* Bl, int ldB,
                         const float* bias, float* C, int ldC, int Mt, int Nt) {
    int blocks = (Mt * Nt * 32 + 255) / 256;
    gemm_split_kernel<<<blocks, 256, 0, stream>>>(A0h, A0l, ld0, K0, A1h, A1l, ld1, K1,
                                                  Bh, Bl, ldB, bias, C, ldC, Mt, Nt);
  };

  // ---- one-time prep ----
  convert_split_kernel<<<cdiv(1536L * 1536, 256), 256, 0, stream>>>(W_ih, WihH, WihL, 1536L * 1536);
  convert_split_kernel<<<cdiv(1536L * 512, 256), 256, 0, stream>>>(W_hh, WhhH, WhhL, 1536L * 512);
  convert_split_kernel<<<cdiv((long)NV * NH, 256), 256, 0, stream>>>(Wo, WoH, WoL, (long)NV * NH);
  convert_split_kernel<<<cdiv((long)NB * NS * NH, 256), 256, 0, stream>>>(enc, encH, encL, (long)NB * NS * NH);
  transpose_split_kernel<<<cdiv(512L * 512, 256), 256, 0, stream>>>(Wa, 512, WaEncTH, WaEncTL);
  gather_embed_kernel<<<cdiv((long)NTS * NB * NE, 256), 256, 0, stream>>>(emb, targets, xsH, xsL);
  h_init_kernel<<<cdiv(NB * NH, 256), 256, 0, stream>>>(hidden, hbuf, hH, hL);

  // enc_proj = enc @ Wa_enc + ba   (hoisted out of the time loop)
  launch_gemm(encH, encL, NH, NH, nullptr, nullptr, 0, 0,
              WaEncTH, WaEncTL, NH, ba, encProj, NH, (NB * NS) / 16, NH / 16);

  // ---- sequential decode: 39 steps ----
  for (int t = 0; t < NTS; ++t) {
    attention_kernel<<<NB, 256, 0, stream>>>(hbuf, Wa, wf, enc, encProj, ctxH, ctxL);
    // gi = [x_t | ctx] @ W_ih^T + b_ih   (A segmented: emb slab + ctx)
    launch_gemm(xsH + (size_t)t * NB * NE, xsL + (size_t)t * NB * NE, NE, NE,
                ctxH, ctxL, NH, NH,
                WihH, WihL, 1536, b_ih, gi, 1536, NB / 16, 1536 / 16);
    // gh = h @ W_hh^T + b_hh
    launch_gemm(hH, hL, NH, NH, nullptr, nullptr, 0, 0,
                WhhH, WhhL, NH, b_hh, gh, 1536, NB / 16, 1536 / 16);
    gru_gate_kernel<<<cdiv(NB * NH, 256), 256, 0, stream>>>(gi, gh, hbuf, hH, hL,
                                                            outputs, outH, outL, t);
  }

  // ---- vocab projection + fused online logsumexp ----
  (void)hipFuncSetAttribute((const void*)vocab_kernel,
                            hipFuncAttributeMaxDynamicSharedMemorySize, 131072);
  vocab_kernel<<<(NROWS / 64) * NCHUNK, 256, 131072, stream>>>(outH, outL, WoH, WoL, bo,
                                                               logp, partM, partS);
  combine_stats_kernel<<<cdiv(NROWS, 256), 256, 0, stream>>>(partM, partS, rowStats);
  logsoftmax_fix_kernel<<<cdiv((long)NROWS * NV / 4, 256), 256, 0, stream>>>(logp, rowStats);
  argmax_kernel<<<NROWS, 128, 0, stream>>>(outputs, amax);
}
